// MultilayerGRU_13829794693274
// MI455X (gfx1250) — matrix operand order repaired and checked by reading, not compiled, not run on hardware
//
#include <hip/hip_runtime.h>
#include <hip/hip_bf16.h>

typedef __attribute__((ext_vector_type(16))) __bf16 v16bf;
typedef __attribute__((ext_vector_type(8)))  float  v8f;

union ABPack {
  v16bf v;
  unsigned short u[16];
  uint4 q[2];
};

__device__ __forceinline__ unsigned short f2bf(float f) {
  unsigned int x = __float_as_uint(f);
  x += 0x7fffu + ((x >> 16) & 1u);          // round to nearest even
  return (unsigned short)(x >> 16);
}
__device__ __forceinline__ float bf2f(unsigned short u) {
  return __uint_as_float((unsigned int)u << 16);
}
__device__ __forceinline__ float sigm(float x) {
  return 1.0f / (1.0f + __expf(-x));
}
__device__ __forceinline__ float tanh_fast(float x) {
  return 1.0f - 2.0f / (1.0f + __expf(2.0f * x));
}

__device__ __forceinline__ v8f wmma_bf16(v16bf a, v16bf b, v8f c) {
  return __builtin_amdgcn_wmma_f32_16x16x32_bf16(false, a, false, b, (short)0, c,
                                                 false, false);
}

// A operand (16x32 bf16) straight from bf16 storage. Lane layout (ISA 7.12.2):
// lanes 0-15 : V0-3 = K k0..k0+7,   V4-7 = K k0+16..k0+23
// lanes 16-31: V0-3 = K k0+8..k0+15, V4-7 = K k0+24..k0+31
__device__ __forceinline__ v16bf load_a_bf(const unsigned short* __restrict__ row,
                                           int k0, int half) {
  ABPack a;
  const int b1 = k0 + half * 8;
  a.q[0] = *(const uint4*)(row + b1);
  a.q[1] = *(const uint4*)(row + b1 + 16);
  return a.v;
}

// A operand computed on the fly: rs = sigmoid(r_pre_f32) * s_bf16
__device__ __forceinline__ v16bf load_a_rs(const float* __restrict__ rrow,
                                           const unsigned short* __restrict__ srow,
                                           int k0, int half) {
  ABPack a;
  const int b1 = k0 + half * 8;
  float4 r0 = *(const float4*)(rrow + b1);
  float4 r1 = *(const float4*)(rrow + b1 + 4);
  float4 r2 = *(const float4*)(rrow + b1 + 16);
  float4 r3 = *(const float4*)(rrow + b1 + 20);
  union { uint4 q; unsigned short u[8]; } s0, s1;
  s0.q = *(const uint4*)(srow + b1);
  s1.q = *(const uint4*)(srow + b1 + 16);
  a.u[0]  = f2bf(sigm(r0.x) * bf2f(s0.u[0]));
  a.u[1]  = f2bf(sigm(r0.y) * bf2f(s0.u[1]));
  a.u[2]  = f2bf(sigm(r0.z) * bf2f(s0.u[2]));
  a.u[3]  = f2bf(sigm(r0.w) * bf2f(s0.u[3]));
  a.u[4]  = f2bf(sigm(r1.x) * bf2f(s0.u[4]));
  a.u[5]  = f2bf(sigm(r1.y) * bf2f(s0.u[5]));
  a.u[6]  = f2bf(sigm(r1.z) * bf2f(s0.u[6]));
  a.u[7]  = f2bf(sigm(r1.w) * bf2f(s0.u[7]));
  a.u[8]  = f2bf(sigm(r2.x) * bf2f(s1.u[0]));
  a.u[9]  = f2bf(sigm(r2.y) * bf2f(s1.u[1]));
  a.u[10] = f2bf(sigm(r2.z) * bf2f(s1.u[2]));
  a.u[11] = f2bf(sigm(r2.w) * bf2f(s1.u[3]));
  a.u[12] = f2bf(sigm(r3.x) * bf2f(s1.u[4]));
  a.u[13] = f2bf(sigm(r3.y) * bf2f(s1.u[5]));
  a.u[14] = f2bf(sigm(r3.z) * bf2f(s1.u[6]));
  a.u[15] = f2bf(sigm(r3.w) * bf2f(s1.u[7]));
  return a.v;
}

// B operand (32x16 bf16) from transposed bf16 weights Wt[n][k] (k contiguous).
// lanes 0-15 = K k0..k0+7,k0+16..k0+23; lanes 16-31 = K k0+8..k0+15,k0+24..k0+31; column n = lane&15.
__device__ __forceinline__ v16bf load_b(const unsigned short* __restrict__ Wt, int n,
                                        int k0, int half) {
  ABPack b;
  const unsigned short* p = Wt + ((size_t)n << 10) + k0 + half * 8;
  b.q[0] = *(const uint4*)(p);
  b.q[1] = *(const uint4*)(p + 16);
  return b.v;
}

// ---------------------------------------------------------------------------
// One-time: fp32 (K x N) row-major weight -> bf16 (N x K) transposed copy.
__global__ __launch_bounds__(256) void transpose_cvt(const float* __restrict__ src,
                                                     unsigned short* __restrict__ dst) {
  __shared__ float tile[32][33];
  const int tx = threadIdx.x;       // 0..31
  const int ty = threadIdx.y;       // 0..7
  const int n0 = blockIdx.x * 32;
  const int k0 = blockIdx.y * 32;
#pragma unroll
  for (int i = 0; i < 32; i += 8)
    tile[ty + i][tx] = src[(size_t)(k0 + ty + i) * 1024 + n0 + tx];
  __syncthreads();
#pragma unroll
  for (int i = 0; i < 32; i += 8)
    dst[(size_t)(n0 + ty + i) * 1024 + k0 + tx] = f2bf(tile[tx][ty + i]);
}

// One-time: (B, T, H) fp32 -> (T, B, H) bf16 permuted cast (for x and h0).
__global__ __launch_bounds__(256) void cvt_perm(const float* __restrict__ src,
                                                unsigned short* __restrict__ dst,
                                                int T) {
  const size_t idx = ((size_t)blockIdx.x * 256 + threadIdx.x) * 4;  // dst index
  const size_t row = idx >> 10;          // t*32 + b
  const int i = (int)(idx & 1023);
  const int t = (int)(row >> 5);
  const int b = (int)(row & 31);
  const float4 f = *(const float4*)(src + ((size_t)b * T + t) * 1024 + i);
  unsigned int lo = (unsigned int)f2bf(f.x) | ((unsigned int)f2bf(f.y) << 16);
  unsigned int hi = (unsigned int)f2bf(f.z) | ((unsigned int)f2bf(f.w) << 16);
  *(uint2*)(dst + idx) = make_uint2(lo, hi);
}

// ---------------------------------------------------------------------------
// Kernel 1 per layer: fused gate pre-activations (all operands bf16):
//   Gpre[0] = inp@Wx + s@Wz + bz   (z_pre)
//   Gpre[1] = inp@Wr + s@Wh + br   (r_pre)
//   Gpre[2] = inp@Wg + bg          (g1)
// 96 waves, each owns a 16(M) x 64(N) tile; K = 1024 (x2 for z,r).
__global__ __launch_bounds__(128) void gru_gates(
    const unsigned short* __restrict__ inp, const unsigned short* __restrict__ st,
    const unsigned short* __restrict__ Wi0, const unsigned short* __restrict__ Wi1,
    const unsigned short* __restrict__ Wi2, const unsigned short* __restrict__ Ws0,
    const unsigned short* __restrict__ Ws1, const float* __restrict__ b0,
    const float* __restrict__ b1, const float* __restrict__ b2,
    float* __restrict__ Gpre) {
  const int wave = (blockIdx.x * 128 + threadIdx.x) >> 5;
  const int lane = threadIdx.x & 31;
  const int g    = wave >> 5;             // 0=z, 1=r, 2=g
  const int rem  = wave & 31;
  const int m0   = (rem >> 4) << 4;       // 0 or 16
  const int n0   = (rem & 15) << 6;       // 0..960
  const int half = lane >> 4;
  const int col  = lane & 15;

  const unsigned short* Wi = (g == 0) ? Wi0 : ((g == 1) ? Wi1 : Wi2);
  const float* bias        = (g == 0) ? b0 : ((g == 1) ? b1 : b2);
  const unsigned short* arow = inp + ((size_t)(m0 + col) << 10);

  v8f acc[4] = {};
  for (int k0 = 0; k0 < 1024; k0 += 32) {
    v16bf a = load_a_bf(arow, k0, half);
#pragma unroll
    for (int ti = 0; ti < 4; ti++) {
      v16bf b = load_b(Wi, n0 + ti * 16 + col, k0, half);
      acc[ti] = wmma_bf16(a, b, acc[ti]);
    }
  }
  if (g < 2) {  // z and r also accumulate the recurrent term s @ W
    const unsigned short* Ws  = (g == 0) ? Ws0 : Ws1;
    const unsigned short* srow = st + ((size_t)(m0 + col) << 10);
    for (int k0 = 0; k0 < 1024; k0 += 32) {
      v16bf a = load_a_bf(srow, k0, half);
#pragma unroll
      for (int ti = 0; ti < 4; ti++) {
        v16bf b = load_b(Ws, n0 + ti * 16 + col, k0, half);
        acc[ti] = wmma_bf16(a, b, acc[ti]);
      }
    }
  }
  // C/D layout: VGPR v of lane -> M = m0 + v + 8*half, N = n0 + ti*16 + col
  float* outg = Gpre + (size_t)g * 32 * 1024;
#pragma unroll
  for (int ti = 0; ti < 4; ti++) {
    const int n    = n0 + ti * 16 + col;
    const float bv = bias[n];
#pragma unroll
    for (int v = 0; v < 8; v++) {
      const int m = m0 + v + half * 8;
      outg[(size_t)m * 1024 + n] = acc[ti][v] + bv;
    }
  }
}

// ---------------------------------------------------------------------------
// Kernel 2 per layer: G2 = (sigmoid(r_pre)*s) @ Whg; epilogue fuses
// ht = sigmoid(z_pre)*s + (1-sigmoid(z_pre))*tanh(g1 + G2).
// ht is stored as bf16 (the only consumer format); optional fp32 copy for d_out.
__global__ __launch_bounds__(128) void gru_update(
    const float* __restrict__ Gpre, const unsigned short* __restrict__ st,
    const unsigned short* __restrict__ Whg, unsigned short* __restrict__ dstbf,
    float* __restrict__ hidout, int hidStride) {
  const int wave = (blockIdx.x * 128 + threadIdx.x) >> 5;
  const int lane = threadIdx.x & 31;
  const int m0   = (wave >> 4) << 4;
  const int n0   = (wave & 15) << 6;
  const int half = lane >> 4;
  const int col  = lane & 15;

  const float* rrow          = Gpre + 32 * 1024 + ((size_t)(m0 + col) << 10);
  const unsigned short* srow = st + ((size_t)(m0 + col) << 10);

  v8f acc[4] = {};
  for (int k0 = 0; k0 < 1024; k0 += 32) {
    v16bf a = load_a_rs(rrow, srow, k0, half);
#pragma unroll
    for (int ti = 0; ti < 4; ti++) {
      v16bf b = load_b(Whg, n0 + ti * 16 + col, k0, half);
      acc[ti] = wmma_bf16(a, b, acc[ti]);
    }
  }
#pragma unroll
  for (int ti = 0; ti < 4; ti++) {
    const int n = n0 + ti * 16 + col;
#pragma unroll
    for (int v = 0; v < 8; v++) {
      const int m    = m0 + v + half * 8;
      const float z  = sigm(Gpre[(size_t)m * 1024 + n]);               // z_pre
      const float sv = bf2f(st[((size_t)m << 10) + n]);
      const float g1 = Gpre[2 * 32 * 1024 + (size_t)m * 1024 + n];     // g1
      const float gg = tanh_fast(g1 + acc[ti][v]);
      const float ht = z * sv + (1.0f - z) * gg;
      dstbf[((size_t)m << 10) + n] = f2bf(ht);
      if (hidout) hidout[(size_t)m * hidStride + n] = ht;
    }
  }
}

// ---------------------------------------------------------------------------
// Hoisted output projection: y[b,t,:] = ht_last(t) @ Wout + bout as one big GEMM,
// M = S*B = 16384, N = K = 1024. 16384 waves of 16x64 tiles.
__global__ __launch_bounds__(256) void out_gemm(const unsigned short* __restrict__ HT,
                                                const unsigned short* __restrict__ Wo,
                                                const float* __restrict__ bout,
                                                float* __restrict__ y) {
  const int wave = (blockIdx.x * 256 + threadIdx.x) >> 5;
  const int lane = threadIdx.x & 31;
  const int m0   = (wave >> 4) << 4;   // row in [0, 16384): m = t*32 + b
  const int n0   = (wave & 15) << 6;
  const int half = lane >> 4;
  const int col  = lane & 15;

  const unsigned short* arow = HT + ((size_t)(m0 + col) << 10);
  v8f acc[4] = {};
  for (int k0 = 0; k0 < 1024; k0 += 32) {
    v16bf a = load_a_bf(arow, k0, half);
#pragma unroll
    for (int ti = 0; ti < 4; ti++) {
      v16bf b = load_b(Wo, n0 + ti * 16 + col, k0, half);
      acc[ti] = wmma_bf16(a, b, acc[ti]);
    }
  }
#pragma unroll
  for (int ti = 0; ti < 4; ti++) {
    const int n    = n0 + ti * 16 + col;
    const float bv = bout[n];
#pragma unroll
    for (int v = 0; v < 8; v++) {
      const int m = m0 + v + half * 8;
      const int t = m >> 5;
      const int b = m & 31;
      y[((size_t)b * 512 + t) * 1024 + n] = acc[ti][v] + bv;
    }
  }
}

// ---------------------------------------------------------------------------
extern "C" void kernel_launch(void* const* d_in, const int* in_sizes, int n_in,
                              void* d_out, int out_size, void* d_ws, size_t ws_size,
                              hipStream_t stream) {
  (void)in_sizes; (void)n_in; (void)out_size; (void)ws_size;
  const float* x    = (const float*)d_in[0];
  const float* h0   = (const float*)d_in[1];
  const float* Wx   = (const float*)d_in[2];
  const float* Wz   = (const float*)d_in[3];
  const float* bz   = (const float*)d_in[4];
  const float* Wr   = (const float*)d_in[5];
  const float* Wh   = (const float*)d_in[6];
  const float* br   = (const float*)d_in[7];
  const float* Wg   = (const float*)d_in[8];
  const float* Whg  = (const float*)d_in[9];
  const float* bg   = (const float*)d_in[10];
  const float* Wout = (const float*)d_in[11];
  const float* bout = (const float*)d_in[12];

  const int H = 1024, Bq = 32, S = 512, L = 3;
  const size_t MAT  = (size_t)H * H;
  const size_t STEP = (size_t)Bq * H;      // one (B,H) state slab

  // workspace: bf16 transposed weights (38MB) + bf16 x (32MB) + bf16 ht log
  // (32MB, doubles as the recurrent-state ring) + small bf16/fp32 temps
  unsigned short* Wxt   = (unsigned short*)d_ws;
  unsigned short* Wzt   = Wxt + 3 * MAT;
  unsigned short* Wrt   = Wzt + 3 * MAT;
  unsigned short* Wht   = Wrt + 3 * MAT;
  unsigned short* Wgt   = Wht + 3 * MAT;
  unsigned short* Whgt  = Wgt + 3 * MAT;
  unsigned short* Wot   = Whgt + 3 * MAT;
  unsigned short* xbf   = Wot + MAT;                 // [S][B][H]
  unsigned short* HTall = xbf + (size_t)S * STEP;    // [S][B][H] last-layer ht
  unsigned short* h0bf  = HTall + (size_t)S * STEP;  // [L][B][H]
  unsigned short* inpA  = h0bf + (size_t)L * STEP;   // layer-0 ht (bf16)
  unsigned short* inpB  = inpA + STEP;               // layer-1 ht (bf16)
  float* Gpre = (float*)(inpB + STEP);               // [3][B][H] fp32

  float* y   = (float*)d_out;                        // (B,S,O)
  float* hid = y + (size_t)Bq * S * H;               // (B,L,H)

  // one-time weight transpose+cvt (results stay L2-resident: 38MB << 192MB L2)
  dim3 tb(32, 8), tg(32, 32);
  for (int l = 0; l < L; l++) {
    transpose_cvt<<<tg, tb, 0, stream>>>(Wx  + l * MAT, Wxt  + l * MAT);
    transpose_cvt<<<tg, tb, 0, stream>>>(Wz  + l * MAT, Wzt  + l * MAT);
    transpose_cvt<<<tg, tb, 0, stream>>>(Wr  + l * MAT, Wrt  + l * MAT);
    transpose_cvt<<<tg, tb, 0, stream>>>(Wh  + l * MAT, Wht  + l * MAT);
    transpose_cvt<<<tg, tb, 0, stream>>>(Wg  + l * MAT, Wgt  + l * MAT);
    transpose_cvt<<<tg, tb, 0, stream>>>(Whg + l * MAT, Whgt + l * MAT);
  }
  transpose_cvt<<<tg, tb, 0, stream>>>(Wout, Wot);
  // one-time activation cvt to bf16 with (B,T,H)->(T,B,H) permute
  cvt_perm<<<(int)((size_t)S * STEP / 4 / 256), 256, 0, stream>>>(x, xbf, S);
  cvt_perm<<<(int)((size_t)L * STEP / 4 / 256), 256, 0, stream>>>(h0, h0bf, L);

  // sequential recurrence: 2 kernels per layer per step, all operands bf16
  for (int t = 0; t < S; t++) {
    for (int j = 0; j < L; j++) {
      const unsigned short* inp =
          (j == 0) ? (xbf + (size_t)t * STEP) : ((j == 1) ? inpA : inpB);

      // reference quirk: state of layer j at step t is last-layer ht from
      // step t+j-L (or h0[:, t+j, :] while t+j < L)
      const int tj = t + j;
      const unsigned short* st = (tj < L) ? (h0bf + (size_t)tj * STEP)
                                          : (HTall + (size_t)(tj - L) * STEP);

      gru_gates<<<24, 128, 0, stream>>>(
          inp, st, Wxt + j * MAT, Wrt + j * MAT, Wgt + j * MAT, Wzt + j * MAT,
          Wht + j * MAT, bz + j * H, br + j * H, bg + j * H, Gpre);

      unsigned short* dstbf =
          (j == 0) ? inpA : ((j == 1) ? inpB : (HTall + (size_t)t * STEP));
      float* hidout = nullptr;
      int hidStride = L * H;
      if (j == L - 1 && t >= S - L) hidout = hid + (size_t)(t - (S - L)) * H;

      gru_update<<<8, 128, 0, stream>>>(Gpre, st, Whgt + j * MAT, dstbf, hidout,
                                        hidStride);
    }
  }

  // one large parallel GEMM for all 512 output projections
  out_gemm<<<2048, 256, 0, stream>>>(HTall, Wot, bout, y);
}
